// MultiHeadSelfAttention_1202590843802
// MI455X (gfx1250) — matrix operand order repaired and checked by reading, not compiled, not run on hardware
//
#include <hip/hip_runtime.h>
#include <hip/hip_bf16.h>

// ---------------------------------------------------------------------------
// Types matching CDNA5 WMMA intrinsic signatures (wave32, 16x16x32 bf16).
// ---------------------------------------------------------------------------
typedef __attribute__((ext_vector_type(16))) __bf16 v16bf;
typedef __attribute__((ext_vector_type(8)))  __bf16 bf16x8;   // 16 bytes
typedef __attribute__((ext_vector_type(8)))  float  v8f;

__device__ __forceinline__ __bf16 f2bf(float f) {
  unsigned u = __builtin_bit_cast(unsigned, f);
  u += 0x7FFFu + ((u >> 16) & 1u);             // round-to-nearest-even
  unsigned short h = (unsigned short)(u >> 16);
  return __builtin_bit_cast(__bf16, h);
}
__device__ __forceinline__ float bf2f(__bf16 b) {
  unsigned short h = __builtin_bit_cast(unsigned short, b);
  unsigned u = ((unsigned)h) << 16;
  return __builtin_bit_cast(float, u);
}

// ---------------------------------------------------------------------------
// Async global -> LDS 16B copy (GLOBAL_LOAD_ASYNC_TO_LDS_B128, ASYNCcnt).
// LDS flat pointers carry the LDS byte offset in addr[31:0] (LDS aperture).
// ---------------------------------------------------------------------------
__device__ __forceinline__ void async_load_b128(void* lds_dst, const void* gsrc) {
  unsigned lds_off = (unsigned)(uintptr_t)lds_dst;
  asm volatile("global_load_async_to_lds_b128 %0, %1, off"
               :: "v"(lds_off), "v"(gsrc)
               : "memory");
}
__device__ __forceinline__ void async_wait0() {
  asm volatile("s_wait_asynccnt 0x0" ::: "memory");
}

// A-fragment (16x32 bf16, MxK): LDS tile stored row-major [M][K], lda elems.
// ISA layout: lane holds row M=lane&15; elems 0..7 -> K in [half*8, half*8+8),
// elems 8..15 -> K in [16+half*8, 16+half*8+8).  Two ds_load_b128 per lane.
__device__ __forceinline__ v16bf lds_load_a(const __bf16* base, int lda, int m0, int k0) {
  int lane = threadIdx.x & 31;
  int m = m0 + (lane & 15), half = lane >> 4;
  const __bf16* p = base + (size_t)m * lda + k0;
  bf16x8 lo = *(const bf16x8*)(p + half * 8);
  bf16x8 hi = *(const bf16x8*)(p + 16 + half * 8);
  v16bf r;
#pragma unroll
  for (int i = 0; i < 8; ++i) { r[i] = lo[i]; r[i + 8] = hi[i]; }
  return r;
}

// B-fragment (32x16 bf16, KxN): LDS tile stored transposed [N][K], ldb elems.
// ISA layout: lane holds column N=lane&15; elems 0..7 -> K in [half*8, half*8+8),
// elems 8..15 -> K in [16+half*8, 16+half*8+8).
__device__ __forceinline__ v16bf lds_load_b(const __bf16* base, int ldb, int n0, int k0) {
  int lane = threadIdx.x & 31;
  int n = n0 + (lane & 15), half = lane >> 4;
  const __bf16* p = base + (size_t)n * ldb + k0;
  bf16x8 lo = *(const bf16x8*)(p + half * 8);
  bf16x8 hi = *(const bf16x8*)(p + 16 + half * 8);
  v16bf r;
#pragma unroll
  for (int i = 0; i < 8; ++i) { r[i] = lo[i]; r[i + 8] = hi[i]; }
  return r;
}

__device__ __forceinline__ v8f wmma_bf16(v16bf a, v16bf b, v8f c) {
  return __builtin_amdgcn_wmma_f32_16x16x32_bf16(false, a, false, b, (short)0, c,
                                                 false, false);
}

// Butterfly max over the 16-lane half this lane lives in (VALU permlane16).
__device__ __forceinline__ float half16_reduce_max(float v) {
#if __has_builtin(__builtin_amdgcn_permlane16)
  unsigned iv;
  iv = __builtin_bit_cast(unsigned, v);   // xor 1
  v = fmaxf(v, __builtin_bit_cast(float, __builtin_amdgcn_permlane16(
                   iv, iv, 0x67452301u, 0xEFCDAB89u, false, false)));
  iv = __builtin_bit_cast(unsigned, v);   // xor 2
  v = fmaxf(v, __builtin_bit_cast(float, __builtin_amdgcn_permlane16(
                   iv, iv, 0x54761032u, 0xDCFE98BAu, false, false)));
  iv = __builtin_bit_cast(unsigned, v);   // xor 4
  v = fmaxf(v, __builtin_bit_cast(float, __builtin_amdgcn_permlane16(
                   iv, iv, 0x32107654u, 0xBA98FEDCu, false, false)));
  iv = __builtin_bit_cast(unsigned, v);   // xor 8
  v = fmaxf(v, __builtin_bit_cast(float, __builtin_amdgcn_permlane16(
                   iv, iv, 0xFEDCBA98u, 0x76543210u, false, false)));
#else
#pragma unroll
  for (int x = 1; x < 16; x <<= 1) v = fmaxf(v, __shfl_xor(v, x, 32));
#endif
  return v;
}

// ---------------------------------------------------------------------------
// Kernel 1: f32 -> bf16 conversion (read-once source: non-temporal loads).
// ---------------------------------------------------------------------------
__global__ __launch_bounds__(256) void cvt_kernel(const float* __restrict__ s,
                                                  __bf16* __restrict__ d, int n) {
  int i = blockIdx.x * 256 + threadIdx.x;
  if (i < n) d[i] = f2bf(__builtin_nontemporal_load(s + i));
}

// ---------------------------------------------------------------------------
// Kernel 2/5: bf16 WMMA GEMM.  C = A(MxK) * B(KxN).
// Block tile 128x128, K-chunk 32, double-buffered LDS, 256 threads = 8 waves;
// each wave 64Mx32N (4x2 tiles of 16x16, f32 accum).  A-tile staging uses
// async global->LDS DMA; B-tile needs a transpose so it stays manual.
// MODE 0: epilogue splits QKV columns into q/k/v [b][h][n][64] bf16 arrays.
// MODE 1: epilogue adds bias, writes fp32 (non-temporal) to out.
// ---------------------------------------------------------------------------
template <int MODE>
__global__ __launch_bounds__(256) void gemm_bf16_kernel(
    const __bf16* __restrict__ A, const __bf16* __restrict__ B, int Kdim, int Ncols,
    __bf16* qb, __bf16* kb, __bf16* vb, float* out, const float* bias) {
  __shared__ __align__(16) __bf16 As[2][128 * 32];
  __shared__ __align__(16) __bf16 Bts[2][128 * 32];   // transposed: [n][k]

  const int tid = threadIdx.x;
  const int lane = tid & 31, wave = tid >> 5;
  const int wm = wave >> 2, wn = wave & 3;
  const int half = lane >> 4, ln = lane & 15;
  const int Mblk = blockIdx.y * 128, Nblk = blockIdx.x * 128;

  auto stage = [&](int k0, int buf) {
    {  // A tile 128x32, row-major: pure 16B/lane copies -> async DMA
      int row = tid >> 1, kh = (tid & 1) * 16;
      const __bf16* src = A + (size_t)(Mblk + row) * Kdim + k0 + kh;
      async_load_b128(&As[buf][row * 32 + kh], src);
      async_load_b128(&As[buf][row * 32 + kh + 8], src + 8);
    }
    {  // B tile 32x128 stored transposed [n][k] (element scatter: manual)
      int kk = tid >> 3, nn = (tid & 7) * 16;
      const __bf16* src = B + (size_t)(k0 + kk) * Ncols + Nblk + nn;
      bf16x8 b0 = *(const bf16x8*)src;
      bf16x8 b1 = *(const bf16x8*)(src + 8);
#pragma unroll
      for (int i = 0; i < 8; ++i) {
        Bts[buf][(nn + i) * 32 + kk]     = b0[i];
        Bts[buf][(nn + 8 + i) * 32 + kk] = b1[i];
      }
    }
  };

  v8f acc[4][2];
#pragma unroll
  for (int mt = 0; mt < 4; ++mt)
#pragma unroll
    for (int nt = 0; nt < 2; ++nt)
#pragma unroll
      for (int i = 0; i < 8; ++i) acc[mt][nt][i] = 0.0f;

  const int nc = Kdim >> 5;
  stage(0, 0);
  async_wait0();
  __syncthreads();

  for (int c = 0; c < nc; ++c) {
    int buf = c & 1;
    if (c + 1 < nc) stage((c + 1) << 5, buf ^ 1);   // overlap with compute

    v16bf af[4];
#pragma unroll
    for (int mt = 0; mt < 4; ++mt)
      af[mt] = lds_load_a(&As[buf][0], 32, wm * 64 + mt * 16, 0);
#pragma unroll
    for (int nt = 0; nt < 2; ++nt) {
      v16bf bf = lds_load_b(&Bts[buf][0], 32, wn * 32 + nt * 16, 0);
#pragma unroll
      for (int mt = 0; mt < 4; ++mt) acc[mt][nt] = wmma_bf16(af[mt], bf, acc[mt][nt]);
    }
    async_wait0();      // our async stores into buf^1 landed in LDS
    __syncthreads();
  }

#pragma unroll
  for (int mt = 0; mt < 4; ++mt)
#pragma unroll
    for (int nt = 0; nt < 2; ++nt)
#pragma unroll
      for (int i = 0; i < 8; ++i) {
        int r = Mblk + wm * 64 + mt * 16 + i + half * 8;
        int c = Nblk + wn * 32 + nt * 16 + ln;
        float val = acc[mt][nt][i];
        if (MODE == 0) {
          int which = c >> 10, hd = c & 1023, h = hd >> 6, d = hd & 63;
          int bb = r >> 11, nrow = r & 2047;
          __bf16* dst = (which == 0) ? qb : ((which == 1) ? kb : vb);
          dst[((size_t)(bb * 16 + h) * 2048 + nrow) * 64 + d] = f2bf(val);
        } else {
          __builtin_nontemporal_store(val + bias[c], &out[(size_t)r * Ncols + c]);
        }
      }
}

// ---------------------------------------------------------------------------
// Kernel 3: RoPE on q and k, interleaved pairs, theta=10000.
// q/k layout: [b*h=32][n=2048][d=64] bf16.  One thread per pair.
// ---------------------------------------------------------------------------
__global__ __launch_bounds__(256) void rope_kernel(__bf16* q, __bf16* k) {
  int tid = blockIdx.x * 256 + threadIdx.x;         // 0 .. 2*2^21-1
  __bf16* arr = (tid >> 21) ? k : q;
  int r = tid & ((1 << 21) - 1);
  int j = r & 31;                                   // pair index (d/2)
  int n = (r >> 5) & 2047;
  int bh = r >> 16;
  size_t idx = ((size_t)bh * 2048 + n) * 64 + 2 * j;
  // inv_freq = 10000^(-(2j)/64) = exp(-j * ln(1e4)/32)
  float inv_freq = __expf(-(float)j * (9.210340371976184f / 32.0f));
  float ang = (float)n * inv_freq;
  float sn, cs;
  __sincosf(ang, &sn, &cs);
  float x0 = bf2f(arr[idx]), x1 = bf2f(arr[idx + 1]);
  arr[idx]     = f2bf(x0 * cs - x1 * sn);
  arr[idx + 1] = f2bf(x1 * cs + x0 * sn);
}

// ---------------------------------------------------------------------------
// Kernel 4: flash attention.  Grid (16 q-blocks, 32 bh); 256 thr = 8 waves.
// Each wave owns 16 query rows; key loop in blocks of 32, double-buffered
// K/V staging (K + Q staged via async DMA).  Row-sum of P on the matrix pipe.
// ---------------------------------------------------------------------------
__global__ __launch_bounds__(256) void attn_kernel(const __bf16* __restrict__ q,
                                                   const __bf16* __restrict__ k,
                                                   const __bf16* __restrict__ v,
                                                   __bf16* __restrict__ o) {
  __shared__ __align__(16) __bf16 Qs[128 * 64];
  __shared__ __align__(16) __bf16 Ks[2][32 * 64];    // [key][d] == Bt for Q*K^T
  __shared__ __align__(16) __bf16 Vts[2][64 * 32];   // [d][key] == Bt for P*V
  __shared__ __align__(16) __bf16 Ps[8][16 * 32];    // per-wave P scratch [m][k]

  const int tid = threadIdx.x, lane = tid & 31, wave = tid >> 5;
  const int half = lane >> 4, ln = lane & 15;
  const int bh = blockIdx.y, qblk = blockIdx.x;
  const size_t base = (size_t)bh * 2048 * 64;
  const float scale = 0.125f;                        // 1/sqrt(64)

  auto stageKV = [&](int kb, int buf) {
    int row = tid >> 3, col = (tid & 7) * 8;
    const __bf16* ks = k + base + (size_t)(kb * 32 + row) * 64 + col;
    async_load_b128(&Ks[buf][row * 64 + col], ks);   // straight copy -> DMA
    const __bf16* vs = v + base + (size_t)(kb * 32 + row) * 64 + col;
    bf16x8 vv = *(const bf16x8*)vs;                  // transpose scatter: manual
#pragma unroll
    for (int i = 0; i < 8; ++i) Vts[buf][(col + i) * 32 + row] = vv[i];
  };

  {  // stage Q block 128x64 (once) via async DMA
    int row = tid >> 1, ch = (tid & 1) * 32;
    const __bf16* src = q + base + (size_t)(qblk * 128 + row) * 64 + ch;
#pragma unroll
    for (int i = 0; i < 4; ++i)
      async_load_b128(&Qs[row * 64 + ch + i * 8], src + i * 8);
  }
  stageKV(0, 0);
  async_wait0();
  __syncthreads();

  // Loop-invariant Q fragments (2 k-steps of 32).
  v16bf aq[2];
  aq[0] = lds_load_a(Qs, 64, wave * 16, 0);
  aq[1] = lds_load_a(Qs, 64, wave * 16, 32);

  // Constant ones B-fragment: P(16x32) @ ones(32x16) -> every C col = rowsum.
  v16bf ones;
  {
    __bf16 one = f2bf(1.0f);
#pragma unroll
    for (int i = 0; i < 16; ++i) ones[i] = one;
  }

  v8f accd[4];
#pragma unroll
  for (int dt = 0; dt < 4; ++dt)
#pragma unroll
    for (int i = 0; i < 8; ++i) accd[dt][i] = 0.0f;
  float rmax[8], rsum[8];
#pragma unroll
  for (int i = 0; i < 8; ++i) { rmax[i] = -3.0e38f; rsum[i] = 0.0f; }

  for (int kb = 0; kb < 64; ++kb) {
    int buf = kb & 1;
    if (kb + 1 < 64) stageKV(kb + 1, buf ^ 1);       // overlap with compute

    // S = Q(16x64) * K^T(64x32): two k-steps, two 16-key tiles
    v8f s0, s1;
#pragma unroll
    for (int i = 0; i < 8; ++i) { s0[i] = 0.0f; s1[i] = 0.0f; }
#pragma unroll
    for (int s = 0; s < 2; ++s) {
      v16bf b0 = lds_load_b(&Ks[buf][0], 64, 0,  s * 32);
      v16bf b1 = lds_load_b(&Ks[buf][0], 64, 16, s * 32);
      s0 = wmma_bf16(aq[s], b0, s0);
      s1 = wmma_bf16(aq[s], b1, s1);
    }

    // online softmax max; row (i + 8*half) spans the 16 lanes of this half
    float p0[8], p1[8], alpha[8];
#pragma unroll
    for (int i = 0; i < 8; ++i) {
      float t0 = s0[i] * scale, t1 = s1[i] * scale;
      float m = half16_reduce_max(fmaxf(t0, t1));
      float mn = fmaxf(rmax[i], m);
      alpha[i] = __expf(rmax[i] - mn);
      rmax[i] = mn;
      p0[i] = __expf(t0 - mn);
      p1[i] = __expf(t1 - mn);
    }
#pragma unroll
    for (int dt = 0; dt < 4; ++dt)
#pragma unroll
      for (int i = 0; i < 8; ++i) accd[dt][i] *= alpha[i];

    // C-layout P -> A-layout via per-wave LDS scratch [16 m][32 key]
#pragma unroll
    for (int i = 0; i < 8; ++i) {
      int rr = i + half * 8;
      Ps[wave][rr * 32 + ln]      = f2bf(p0[i]);
      Ps[wave][rr * 32 + 16 + ln] = f2bf(p1[i]);
    }
    __builtin_amdgcn_wave_barrier();
    asm volatile("s_wait_dscnt 0x0" ::: "memory");   // LDS RAW across lanes

    v16bf pa = lds_load_a(&Ps[wave][0], 32, 0, 0);

    // Row-sum on the matrix pipe: rs[i] = sum_k P[row i][k]
    v8f zero;
#pragma unroll
    for (int i = 0; i < 8; ++i) zero[i] = 0.0f;
    v8f rs = wmma_bf16(pa, ones, zero);

#pragma unroll
    for (int dt = 0; dt < 4; ++dt) {
      v16bf bv = lds_load_b(&Vts[buf][0], 32, dt * 16, 0);
      accd[dt] = wmma_bf16(pa, bv, accd[dt]);
    }
#pragma unroll
    for (int i = 0; i < 8; ++i) rsum[i] = rsum[i] * alpha[i] + rs[i];

    async_wait0();     // async K stage for kb+1 landed
    __syncthreads();   // + everyone done reading buf
  }

  // epilogue: O /= rowsum, write bf16 to o[b][n][h*64+d]
  int b = bh >> 4, h = bh & 15;
#pragma unroll
  for (int i = 0; i < 8; ++i) {
    float inv = 1.0f / rsum[i];
    int n = qblk * 128 + wave * 16 + i + half * 8;
    size_t rowbase = ((size_t)b * 2048 + n) * 1024 + h * 64;
#pragma unroll
    for (int dt = 0; dt < 4; ++dt)
      o[rowbase + dt * 16 + ln] = f2bf(accd[dt][i] * inv);
  }
}

// ---------------------------------------------------------------------------
// Host-side launch.
// ---------------------------------------------------------------------------
extern "C" void kernel_launch(void* const* d_in, const int* in_sizes, int n_in,
                              void* d_out, int out_size, void* d_ws, size_t ws_size,
                              hipStream_t stream) {
  const float* x    = (const float*)d_in[0];  // (2,2048,1024)
  const float* wqkv = (const float*)d_in[1];  // (1024,3072)
  const float* wout = (const float*)d_in[2];  // (1024,1024)
  const float* bout = (const float*)d_in[3];  // (1024,)
  float* out = (float*)d_out;                 // (2,2048,1024) fp32

  const size_t NX    = (size_t)4096 * 1024;   // x / o elems
  const size_t NWQKV = (size_t)1024 * 3072;
  const size_t NWOUT = (size_t)1024 * 1024;
  const size_t NHEAD = (size_t)32 * 2048 * 64;

  __bf16* p = (__bf16*)d_ws;
  __bf16* xb    = p; p += NX;
  __bf16* wqkvb = p; p += NWQKV;
  __bf16* woutb = p; p += NWOUT;
  __bf16* qb    = p; p += NHEAD;
  __bf16* kb    = p; p += NHEAD;
  __bf16* vb    = p; p += NHEAD;
  __bf16* ob    = p; p += NX;
  (void)ws_size; (void)n_in; (void)in_sizes; (void)out_size;

  cvt_kernel<<<(int)((NX + 255) / 256), 256, 0, stream>>>(x, xb, (int)NX);
  cvt_kernel<<<(int)((NWQKV + 255) / 256), 256, 0, stream>>>(wqkv, wqkvb, (int)NWQKV);
  cvt_kernel<<<(int)((NWOUT + 255) / 256), 256, 0, stream>>>(wout, woutb, (int)NWOUT);

  // QKV projection: (4096x1024)x(1024x3072), split heads in epilogue
  gemm_bf16_kernel<0><<<dim3(24, 32), 256, 0, stream>>>(
      xb, wqkvb, 1024, 3072, qb, kb, vb, nullptr, nullptr);

  // RoPE on q and k: 2 * 32 * 2048 * 32 pairs
  rope_kernel<<<16384, 256, 0, stream>>>(qb, kb);

  // Flash attention: (16 q-blocks) x (32 bh)
  attn_kernel<<<dim3(16, 32), 256, 0, stream>>>(qb, kb, vb, ob);

  // Output projection + bias: (4096x1024)x(1024x1024) -> fp32
  gemm_bf16_kernel<1><<<dim3(8, 32), 256, 0, stream>>>(
      ob, woutb, 1024, 1024, nullptr, nullptr, nullptr, out, bout);
}
